// HiddenMarkovModel_58892591563425
// MI455X (gfx1250) — matrix operand order repaired and checked by reading, not compiled, not run on hardware
//
#include <hip/hip_runtime.h>
#include <hip/hip_bf16.h>
#include <math.h>

// HMM forward logZ on MI455X (gfx1250, wave32, WMMA).
//   K=512 tags, D=512, V=vocab (50000), L=words (4096).
// Pipeline:
//   k_softmaxA : A = softmax(WA with col BOS = -inf) + TINY          [K,K] f32
//   k_rowstats : sumexp[k] = sum_v exp(ThetaB[k]·E[v])   (bf16 WMMA, streamed;
//                kk-outer / 4-tile-inner, raw v16f B loads double-buffered so
//                global_load_b128s overlap the WMMA chain instead of
//                serializing on s_wait_loadcnt 0)
//   k_sel      : logBsel[i,t] = log(exp(ThetaB[t]·E[w_i] - log sumexp[t]) + TINY)
//   k_scan     : alpha recursion as prob-space matvec with exact max-shift;
//                32 persistent WGs, A-slice (512x16 = 32KB) in LDS,
//                device-wide sense barrier (atomics + s_sleep) per step.

#define KTAGS 512
#define DDIM  512
#define BOSID 510
#define EOSID 511
#define TINYF 1e-45f
#define LOG_TINY (-103.2789307f)   // logf(1e-45f as float == 1.4013e-45)
#define SCAN_BLOCKS 32

typedef __attribute__((ext_vector_type(16))) __bf16 v16bf;
typedef __attribute__((ext_vector_type(8)))  float  v8f;
typedef __attribute__((ext_vector_type(16))) float  v16f;

// ---------------- block reductions (blockDim.x == 512, 16 waves) -----------
__device__ __forceinline__ float blockMax512(float v, float* sm) {
#pragma unroll
  for (int o = 16; o >= 1; o >>= 1) v = fmaxf(v, __shfl_xor(v, o, 32));
  if ((threadIdx.x & 31) == 0) sm[threadIdx.x >> 5] = v;
  __syncthreads();
  float r = sm[0];
#pragma unroll
  for (int i = 1; i < 16; ++i) r = fmaxf(r, sm[i]);
  __syncthreads();
  return r;
}

__device__ __forceinline__ float blockSum512(float v, float* sm) {
#pragma unroll
  for (int o = 16; o >= 1; o >>= 1) v += __shfl_xor(v, o, 32);
  if ((threadIdx.x & 31) == 0) sm[threadIdx.x >> 5] = v;
  __syncthreads();
  float r = sm[0];
#pragma unroll
  for (int i = 1; i < 16; ++i) r += sm[i];
  __syncthreads();
  return r;
}

// ---------------- pass 1: A = softmax(WA masked) + TINY --------------------
__global__ void k_softmaxA(const float* __restrict__ WA, float* __restrict__ A) {
  __shared__ float sm[16];
  const int k = blockIdx.x;
  const int t = threadIdx.x;
  float wa = (t == BOSID) ? -__builtin_inff() : WA[k * KTAGS + t];
  float m = blockMax512(wa, sm);
  float e = expf(wa - m);            // exp(-inf) == 0 for the BOS column
  float s = blockSum512(e, sm);
  A[k * KTAGS + t] = e / s + TINYF;
}

// ---------------- WMMA fragment helpers (f32 raw -> bf16 in-register) ------
// A-frag (16x32, bf16, contiguous k order (not AMD's table)): lane l holds M = l&15; element e maps to
// K = e + 16*(l>>4)  -> two contiguous 8-wide (32B) vector loads.
__device__ __forceinline__ v16bf load_a_frag(const float* __restrict__ ThetaB,
                                             int m0, int kk, int nl, int half) {
  const float* pa = ThetaB + (size_t)(m0 + nl) * DDIM + kk + 16 * half;
  const v8f lo = *(const v8f*)(pa);
  const v8f hi = *(const v8f*)(pa + 8);
  v16bf a;
#pragma unroll
  for (int e = 0; e < 8; ++e) {
    a[e]     = (__bf16)lo[e];
    a[8 + e] = (__bf16)hi[e];
  }
  return a;
}

// B-frag (32x16): lane l holds column N = l&15, K = e + 16*(l>>4)
// -> one 64B raw vector (4x global_load_b128 into distinct regs, one wait).
__device__ __forceinline__ v16f load_b_raw(const float* __restrict__ Erow,
                                           int kk, int half) {
  return *(const v16f*)(Erow + kk + 16 * half);
}

__device__ __forceinline__ v16bf cvt_b(v16f r) {
  v16bf b;
#pragma unroll
  for (int e = 0; e < 16; ++e) b[e] = (__bf16)r[e];
  return b;
}

// ---------------- pass 2: row sum-of-exp over full V (streamed) ------------
// grid = (ceil(NT/16), K/16), block = 128 (4 waves); each wave owns 4 n-tiles
// (32 VGPRs of accumulators). kk outer: each A-fragment feeds 4 WMMAs; the
// raw B vector for tile j+1 is issued before tile j's cvt+WMMA (explicit MLP).
__global__ void k_rowstats(const float* __restrict__ ThetaB,
                           const float* __restrict__ E,
                           float* __restrict__ sumexp, int NT) {
  const int lane = threadIdx.x & 31;
  const int wv   = threadIdx.x >> 5;
  const int m0   = blockIdx.y * 16;
  const int half = lane >> 4;
  const int nl   = lane & 15;
  const int tbase = (blockIdx.x * 4 + wv) * 4;

  // Clamp out-of-range tiles to a valid row (reads are safe, contribution
  // masked later) so EXEC stays all-ones for every WMMA.
  const float* Erow[4];
#pragma unroll
  for (int j = 0; j < 4; ++j) {
    int tile = tbase + j;
    if (tile >= NT) tile = NT - 1;
    Erow[j] = E + (size_t)(tile * 16 + nl) * DDIM;
  }

  v8f c[4];
#pragma unroll
  for (int j = 0; j < 4; ++j) c[j] = (v8f){};

  for (int kk = 0; kk < DDIM; kk += 32) {
    const v16bf a = load_a_frag(ThetaB, m0, kk, nl, half);
    v16f raw = load_b_raw(Erow[0], kk, half);
#pragma unroll
    for (int j = 0; j < 4; ++j) {
      v16f nxt = raw;
      if (j + 1 < 4) nxt = load_b_raw(Erow[j + 1], kk, half);  // overlap
      const v16bf b = cvt_b(raw);
      c[j] = __builtin_amdgcn_wmma_f32_16x16x32_bf16(false, a, false, b,
                                                     (short)0, c[j], false, false);
      raw = nxt;
    }
  }

  float acc[8];
#pragma unroll
  for (int r = 0; r < 8; ++r) acc[r] = 0.f;
#pragma unroll
  for (int j = 0; j < 4; ++j) {
    if (tbase + j < NT) {
#pragma unroll
      for (int r = 0; r < 8; ++r) acc[r] += expf(c[j][r]);
    }
  }

  // reduce over the 16 N-lanes of each half; C-frag row M = r + 8*half
#pragma unroll
  for (int r = 0; r < 8; ++r) {
    float v = acc[r];
    v += __shfl_xor(v, 1, 32);
    v += __shfl_xor(v, 2, 32);
    v += __shfl_xor(v, 4, 32);
    v += __shfl_xor(v, 8, 32);
    if (nl == 0) atomicAdd(&sumexp[m0 + r + 8 * half], v);
  }
}

// ---------------- pass 3: logB at the gathered word columns ----------------
// grid = (ceil((L/16)/4), K/16), block = 128; one 16-col tile per wave.
__global__ void k_sel(const float* __restrict__ ThetaB,
                      const float* __restrict__ E,
                      const int* __restrict__ words,
                      const float* __restrict__ sumexp,
                      float* __restrict__ logBsel, int L) {
  const int lane = threadIdx.x & 31;
  const int wv   = threadIdx.x >> 5;
  const int m0   = blockIdx.y * 16;
  const int half = lane >> 4;
  const int nl   = lane & 15;
  const int n0   = (blockIdx.x * 4 + wv) * 16;  // wave-uniform
  if (n0 >= L) return;

  const int wrd = words[n0 + nl];
  const float* __restrict__ Erow = E + (size_t)wrd * DDIM;
  v8f c = {};
  v16f raw = load_b_raw(Erow, 0, half);
#pragma unroll
  for (int kk = 0; kk < DDIM; kk += 32) {
    v16f nxt = raw;
    if (kk + 32 < DDIM) nxt = load_b_raw(Erow, kk + 32, half);  // overlap
    const v16bf a = load_a_frag(ThetaB, m0, kk, nl, half);
    const v16bf b = cvt_b(raw);
    c = __builtin_amdgcn_wmma_f32_16x16x32_bf16(false, a, false, b,
                                                (short)0, c, false, false);
    raw = nxt;
  }
#pragma unroll
  for (int r = 0; r < 8; ++r) {
    const int m = m0 + r + 8 * half;
    float lb;
    if (m >= BOSID) {                        // BOS and EOS rows: B := 0 + TINY
      lb = LOG_TINY;
    } else {
      lb = logf(expf(c[r] - logf(sumexp[m])) + TINYF);
    }
    logBsel[(size_t)(n0 + nl) * KTAGS + m] = lb;   // [i][t] layout for scan
  }
}

// ---------------- device-wide sense barrier (cumulative counting) ----------
__device__ __forceinline__ void grid_barrier(unsigned* cnt,
                                             volatile unsigned* rel,
                                             unsigned target) {
  __threadfence();                 // release: make alpha stores visible
  __syncthreads();
  if (threadIdx.x == 0) {
    unsigned prev = atomicAdd(cnt, 1u);
    if (prev + 1u == (unsigned)SCAN_BLOCKS * target) {
      atomicExch((unsigned*)rel, target);
    }
    while (*rel < target) __builtin_amdgcn_s_sleep(2);
  }
  __syncthreads();
  __threadfence();                 // acquire: invalidate near caches
}

// ---------------- pass 4: forward recursion ---------------------------------
// alpha'[t] = m + log( sum_s exp(alpha[s]-m) * A[s,t] ) + logB[t, w_i]
// 32 WGs x 512 threads; WG holds A columns [c0, c0+16) in LDS (32 KB);
// wave w owns column c0+w: 16 MACs/lane + wave reduce per step.
__global__ void k_scan(const float* __restrict__ A,
                       const float* __restrict__ logBsel,
                       float* __restrict__ alphaBuf,
                       unsigned* __restrict__ barCnt,
                       unsigned* __restrict__ barRel,
                       float* __restrict__ out, int L) {
  __shared__ float Asl[16 * KTAGS];    // [colLocal][s]
  __shared__ float vsh[KTAGS];
  __shared__ float sm[16];
  const int tid  = threadIdx.x;
  const int lane = tid & 31;
  const int wv   = tid >> 5;
  const int c0   = blockIdx.x * 16;

  // preload this WG's A slice: Asl[c*512 + s] = A[s*512 + c0 + c]
  for (int idx = tid; idx < 16 * KTAGS; idx += 512) {
    const int c = idx & 15;
    const int s = idx >> 4;
    Asl[c * KTAGS + s] = A[(size_t)s * KTAGS + c0 + c];
  }

  float* alpha0 = alphaBuf;
  float* alpha1 = alphaBuf + KTAGS;
  if (blockIdx.x == 0) alpha0[tid] = (tid == BOSID) ? 0.f : -__builtin_inff();

  unsigned target = 1;
  grid_barrier(barCnt, barRel, target++);   // alpha0 initialized everywhere

  for (int i = 0; i < L; ++i) {
    const float* cur = (i & 1) ? alpha1 : alpha0;
    float*       nxt = (i & 1) ? alpha0 : alpha1;

    const float a = cur[tid];
    const float m = blockMax512(a, sm);
    vsh[tid] = expf(a - m);
    __syncthreads();

    float p = 0.f;
    const float* Ac = Asl + wv * KTAGS;
#pragma unroll
    for (int k2 = 0; k2 < 16; ++k2) {
      const int s = lane + 32 * k2;
      p += vsh[s] * Ac[s];
    }
#pragma unroll
    for (int o = 16; o >= 1; o >>= 1) p += __shfl_xor(p, o, 32);

    if (lane == 0) {
      nxt[c0 + wv] = m + logf(p) + logBsel[(size_t)i * KTAGS + c0 + wv];
      if (i + 1 < L)
        __builtin_prefetch(&logBsel[(size_t)(i + 1) * KTAGS + c0 + wv], 0, 1);
    }
    grid_barrier(barCnt, barRel, target++);
  }

  if (blockIdx.x == 0) {
    const float* fin = (L & 1) ? alpha1 : alpha0;
    // faithful to reference: logsumexp(alpha_last + A[:, EOS]) with A = probs
    const float x  = fin[tid] + A[(size_t)tid * KTAGS + EOSID];
    const float m2 = blockMax512(x, sm);
    const float s2 = blockSum512(expf(x - m2), sm);
    if (tid == 0) out[0] = m2 + logf(s2);
  }
}

// ---------------------------------------------------------------------------
extern "C" void kernel_launch(void* const* d_in, const int* in_sizes, int n_in,
                              void* d_out, int out_size, void* d_ws, size_t ws_size,
                              hipStream_t stream) {
  const float* ThetaB = (const float*)d_in[0];   // [K, D]
  const float* WA     = (const float*)d_in[1];   // [K, K]
  const float* E      = (const float*)d_in[2];   // [V, D]
  const int*   words  = (const int*)d_in[3];     // [L]
  const int V = in_sizes[2] / DDIM;
  const int L = in_sizes[3];
  float* out = (float*)d_out;

  // workspace carve-out (256B aligned slabs)
  char* ws = (char*)d_ws;
  size_t off = 0;
  auto carve = [&](size_t bytes) -> void* {
    void* p = ws + off;
    off = (off + bytes + 255) & ~(size_t)255;
    return p;
  };
  float*    A       = (float*)carve((size_t)KTAGS * KTAGS * sizeof(float));
  float*    logBsel = (float*)carve((size_t)L * KTAGS * sizeof(float));
  float*    sumexp  = (float*)carve(KTAGS * sizeof(float));
  float*    alphaB  = (float*)carve(2 * KTAGS * sizeof(float));
  unsigned* barCnt  = (unsigned*)carve(256);
  unsigned* barRel  = (unsigned*)carve(256);

  // per-call reset (graph-capture safe)
  hipMemsetAsync(sumexp, 0, KTAGS * sizeof(float), stream);
  hipMemsetAsync(barCnt, 0, 256, stream);
  hipMemsetAsync(barRel, 0, 256, stream);

  // pass 1: transition probabilities A
  k_softmaxA<<<KTAGS, 512, 0, stream>>>(WA, A);

  // pass 2: streamed WMMA GEMM over full V for softmax denominators
  const int NT = V / 16;                      // 16-col tiles (V divisible by 16)
  dim3 g1((NT + 15) / 16, KTAGS / 16);        // 4 waves x 4 tiles per block
  k_rowstats<<<g1, 128, 0, stream>>>(ThetaB, E, sumexp, NT);

  // pass 3: WMMA GEMM over the gathered word columns -> logBsel
  dim3 g2(((L / 16) + 3) / 4, KTAGS / 16);
  k_sel<<<g2, 128, 0, stream>>>(ThetaB, E, words, sumexp, logBsel, L);

  // pass 4: persistent multi-WG scan with device-wide barrier
  k_scan<<<SCAN_BLOCKS, 512, 0, stream>>>(A, logBsel, alphaB, barCnt, barRel,
                                          out, L);
}
